// AttentionDecoder_15848429322545
// MI455X (gfx1250) — compile-verified, matrix operand order checked by reading, not run on hardware
//
#include <hip/hip_runtime.h>
#include <math.h>

// ---------------- problem constants ----------------
#define DIM      768
#define DEPTH    8
#define HEADS    12
#define DHEAD    64
#define MLP      3072
#define NTOK     1024      // H*W = 32*32
#define BATCH    4
#define FDIM     256
#define NROWS    (BATCH * NTOK)   // 4096
#define ATTN_SCALE 0.125f         // 64^-0.5
#define LN_EPS   1e-5f

// ---------------- CDNA5 WMMA / TDM types ----------------
typedef __bf16 bf16_t;
typedef __attribute__((ext_vector_type(16))) bf16_t         bf16x16;
typedef __attribute__((ext_vector_type(8)))  float          f32x8;
typedef __attribute__((ext_vector_type(8)))  unsigned short u16x8;
typedef __attribute__((ext_vector_type(4)))  unsigned int   u32x4;
typedef __attribute__((ext_vector_type(8)))  int            i32x8;
typedef __attribute__((ext_vector_type(4)))  int            i32x4;

__device__ __forceinline__ unsigned short f2bf(float f) {
  unsigned int u = __float_as_uint(f);
  unsigned int r = u + 0x7fffu + ((u >> 16) & 1u);
  return (unsigned short)(r >> 16);
}

union FragU { u16x8 h[2]; bf16x16 f; };

// CDNA5 16-bit A/B fragment pattern: elems 0..7 = k in [klo,klo+8), 8..15 = [khi,khi+8)
__device__ __forceinline__ bf16x16 load_frag_lds(const unsigned short* base, int stride,
                                                 int row, int klo, int khi) {
  FragU u;
  u.h[0] = *(const u16x8*)(base + row * stride + klo);
  u.h[1] = *(const u16x8*)(base + row * stride + khi);
  return u.f;
}

__device__ __forceinline__ f32x8 wmma_bf16(bf16x16 a, bf16x16 b, f32x8 c) {
  return __builtin_amdgcn_wmma_f32_16x16x32_bf16(false, a, false, b, (short)0, c, false, false);
}

// ---------------- Tensor Data Mover: 2D bf16 tile -> padded LDS ----------------
// Copies tile_d1 rows x tile_d0 bf16 elements (row stride `stride0` elements in
// global) into LDS at lds_addr. pad_enable inserts 4 DWORDs (16B) after every
// 32 DWORDs (128B) stored -> LDS row pitch 144B = 72 shorts (matches our tiles,
// tile_d0 must be 64).
__device__ __forceinline__ void tdm_load_2d(unsigned lds_addr, const void* gaddr,
                                            unsigned tile_d0, unsigned tile_d1,
                                            unsigned stride0) {
  unsigned long long ga = (unsigned long long)gaddr;
  u32x4 g0;
  g0[0] = 1u;                                              // count=1 (valid), user mode
  g0[1] = lds_addr;                                        // D#.lds_addr
  g0[2] = (unsigned)(ga & 0xffffffffu);                    // global_addr[31:0]
  g0[3] = (unsigned)((ga >> 32) & 0x01ffffffu) | (2u << 30); // addr[56:32] | type=2
  i32x8 g1;
  g1[0] = (int)((1u << 16)      // data_size = 1 (2 bytes)
              | (1u << 20)      // pad_enable
              | (4u << 22)      // pad_interval: 32 DWORDs
              | (3u << 25));    // pad_amount: 4 DWORDs
  g1[1] = (int)((tile_d0 & 0xffffu) << 16);                // tensor_dim0[15:0] (== tile)
  g1[2] = (int)(((tile_d0 >> 16) & 0xffffu) | ((tile_d1 & 0xffffu) << 16)); // d0 hi | tensor_dim1 lo
  g1[3] = (int)(((tile_d1 >> 16) & 0xffffu) | (tile_d0 << 16));             // d1 hi | tile_dim0
  g1[4] = (int)(tile_d1 & 0xffffu);                        // tile_dim1 (tile_dim2 = 0)
  g1[5] = (int)stride0;                                    // tensor_dim0_stride[31:0]
  g1[6] = 0;                                               // stride0[47:32] | stride1[15:0]
  g1[7] = 0;
  i32x4 z4 = {0, 0, 0, 0};
  i32x8 z8 = {0, 0, 0, 0, 0, 0, 0, 0};
  __builtin_amdgcn_tensor_load_to_lds(g0, g1, z4, z4, z8, 0);
}

__device__ __forceinline__ unsigned lds_off(const void* p) {
  return (unsigned)(unsigned long long)p;   // generic LDS ptr: low 32 bits = LDS offset
}

// ---------------- sine positional embedding (normalize=True, scale=2pi) ----------------
__device__ __forceinline__ float pos_embed(int n, int d) {
  const float two_pi = 6.283185307179586f;
  int   dd    = (d < 384) ? d : d - 384;
  float coord = (d < 384) ? (float)((n >> 5) + 1) : (float)((n & 31) + 1);
  coord = coord / (32.0f + 1e-6f) * two_pi;
  float dimt = __powf(10000.0f, (float)(2 * (dd >> 1)) * (1.0f / 384.0f));
  float p = coord / dimt;
  return (dd & 1) ? __cosf(p) : __sinf(p);
}

// ---------------- GEMM: out[M,N] = X[M,K](bf16) @ Wt[N,K](bf16)^T (+epilogue) ----------------
enum { EPI_NONE = 0, EPI_BIAS = 1, EPI_BIAS_RESID = 2, EPI_BIAS_GELU = 3, EPI_BIAS_POS = 4 };

template <int EPI, bool OUTBF>
__global__ __launch_bounds__(256) void gemm_kernel(
    const unsigned short* __restrict__ X,   // bf16 [M][ldx]
    const unsigned short* __restrict__ Wt,  // bf16 [N][K] row-major (pre-transposed weights)
    const float* __restrict__ bias, const float* __restrict__ resid,
    void* __restrict__ outp, int M, int N, int K, int ldx, int ldo) {
  __shared__ __align__(16) unsigned short As[64 * 72];    // 64 rows x 64 k, TDM-padded to 72
  __shared__ __align__(16) unsigned short Bs[128 * 72];   // 128 cols x 64 k, TDM-padded to 72
  const int t     = threadIdx.x;
  const int lane  = t & 31;
  const int wv    = t >> 5;
  const int waveM = wv >> 2;
  const int waveN = wv & 3;
  const int half  = lane >> 4;
  const int lr    = lane & 15;
  const int bm    = blockIdx.y * 64;
  const int bn    = blockIdx.x * 128;

  const unsigned asOff = lds_off(As);
  const unsigned bsOff = lds_off(Bs);

  f32x8 acc[2][2] = {};

  for (int k0 = 0; k0 < K; k0 += 64) {
    if (wv == 0) {   // one wave drives the Tensor Data Mover for both tiles
      tdm_load_2d(asOff, X + (size_t)bm * ldx + k0, 64, 64, (unsigned)ldx);
      tdm_load_2d(bsOff, Wt + (size_t)bn * K + k0, 64, 128, (unsigned)K);
      __builtin_amdgcn_s_wait_tensorcnt(0);
    }
    __syncthreads();
#pragma unroll
    for (int ks = 0; ks < 64; ks += 32) {
      const int klo = ks + half * 8;
      const int khi = ks + 16 + half * 8;
      bf16x16 a0 = load_frag_lds(As, 72, waveM * 32 + lr,      klo, khi);
      bf16x16 a1 = load_frag_lds(As, 72, waveM * 32 + 16 + lr, klo, khi);
      bf16x16 b0 = load_frag_lds(Bs, 72, waveN * 32 + lr,      klo, khi);
      bf16x16 b1 = load_frag_lds(Bs, 72, waveN * 32 + 16 + lr, klo, khi);
      acc[0][0] = wmma_bf16(a0, b0, acc[0][0]);
      acc[0][1] = wmma_bf16(a0, b1, acc[0][1]);
      acc[1][0] = wmma_bf16(a1, b0, acc[1][0]);
      acc[1][1] = wmma_bf16(a1, b1, acc[1][1]);
    }
    __syncthreads();
  }

  float* outf = (float*)outp;
  unsigned short* outb = (unsigned short*)outp;
#pragma unroll
  for (int i = 0; i < 2; ++i) {
#pragma unroll
    for (int j = 0; j < 2; ++j) {
      const int col = bn + waveN * 32 + j * 16 + lr;
#pragma unroll
      for (int v = 0; v < 8; ++v) {
        const int row = bm + waveM * 32 + i * 16 + v + 8 * half;
        float val = acc[i][j][v];
        if (EPI != EPI_NONE) val += bias[col];
        if (EPI == EPI_BIAS_GELU)  val = 0.5f * val * (1.0f + erff(val * 0.7071067811865475f));
        if (EPI == EPI_BIAS_RESID) val += resid[(size_t)row * ldo + col];
        if (EPI == EPI_BIAS_POS)   val += pos_embed(row & (NTOK - 1), col);
        if (OUTBF) outb[(size_t)row * ldo + col] = f2bf(val);
        else       outf[(size_t)row * ldo + col] = val;
      }
    }
  }
}

// ---------------- flash attention: one WG = (b, head, 128 query rows) ----------------
__global__ __launch_bounds__(256) void attn_kernel(const unsigned short* __restrict__ qkv,
                                                   unsigned short* __restrict__ o) {
  __shared__ __align__(16) unsigned short Ks[32 * 72];      // [key][feat64] TDM-padded
  __shared__ __align__(16) unsigned short Vt[64 * 40];      // [feat][key32] pad->40
  __shared__ __align__(16) unsigned short Ps[8 * 16 * 40];  // per-wave P tile [row][key]
  const int t    = threadIdx.x;
  const int lane = t & 31;
  const int wv   = t >> 5;
  const int half = lane >> 4;
  const int lr   = lane & 15;
  const int b    = blockIdx.z;
  const int hh   = blockIdx.y;
  const int row0 = blockIdx.x * 128 + wv * 16;
  const unsigned short* base = qkv + (size_t)b * NTOK * (3 * DIM) + hh * DHEAD;
  const unsigned ksOff = lds_off(Ks);

  // Q fragments (bf16 direct): A-layout, K = 64 feats -> 2 frags
  bf16x16 aq[2];
  {
    const unsigned short* qr = base + (size_t)(row0 + lr) * (3 * DIM);
#pragma unroll
    for (int kf = 0; kf < 2; ++kf) {
      FragU u;
      u.h[0] = *(const u16x8*)(qr + kf * 32 + half * 8);
      u.h[1] = *(const u16x8*)(qr + kf * 32 + 16 + half * 8);
      aq[kf] = u.f;
    }
  }

  float mrow[8], lrow[8];
  f32x8 oacc[4] = {};
#pragma unroll
  for (int v = 0; v < 8; ++v) { mrow[v] = -1e30f; lrow[v] = 0.0f; }

  const int vf = t & 63;        // feature for V staging
  const int vkg = t >> 6;       // key group (8 keys)

  for (int kt = 0; kt < NTOK; kt += 32) {
    // K tile: 32 rows x 64 feats, row stride 3*DIM -> TDM
    if (wv == 0) {
      tdm_load_2d(ksOff, base + (size_t)kt * (3 * DIM) + DIM, 64, 32, 3 * DIM);
      __builtin_amdgcn_s_wait_tensorcnt(0);
    }
    // V tile transposed into LDS: gather 8 keys of one feature, store b128
    {
      u16x8 tmp;
#pragma unroll
      for (int e = 0; e < 8; ++e)
        tmp[e] = base[(size_t)(kt + vkg * 8 + e) * (3 * DIM) + 2 * DIM + vf];
      *(u16x8*)&Vt[vf * 40 + vkg * 8] = tmp;
    }
    __syncthreads();

    // S = Q @ K^T for 32 keys: two 16x16 D tiles
    f32x8 s[2];
#pragma unroll
    for (int jt = 0; jt < 2; ++jt) {
      bf16x16 bk0 = load_frag_lds(Ks, 72, jt * 16 + lr, half * 8,      16 + half * 8);
      bf16x16 bk1 = load_frag_lds(Ks, 72, jt * 16 + lr, 32 + half * 8, 48 + half * 8);
      f32x8 z = {};
      z = wmma_bf16(aq[0], bk0, z);
      z = wmma_bf16(aq[1], bk1, z);
#pragma unroll
      for (int v = 0; v < 8; ++v) s[jt][v] = z[v] * ATTN_SCALE;
    }

    // online softmax; row r = v + 8*half spans 16 lanes of this half
    float cs[8];
#pragma unroll
    for (int v = 0; v < 8; ++v) {
      float pm = fmaxf(s[0][v], s[1][v]);
#pragma unroll
      for (int off = 1; off < 16; off <<= 1) pm = fmaxf(pm, __shfl_xor(pm, off, 32));
      float mnew = fmaxf(mrow[v], pm);
      cs[v]   = __expf(mrow[v] - mnew);
      mrow[v] = mnew;
      float p0 = __expf(s[0][v] - mnew);
      float p1 = __expf(s[1][v] - mnew);
      s[0][v] = p0; s[1][v] = p1;
      float r = p0 + p1;
#pragma unroll
      for (int off = 1; off < 16; off <<= 1) r += __shfl_xor(r, off, 32);
      lrow[v] = lrow[v] * cs[v] + r;
    }

    // relayout P (D-layout -> A-layout) via per-wave LDS slice
    unsigned short* pw = Ps + wv * (16 * 40);
#pragma unroll
    for (int jt = 0; jt < 2; ++jt)
#pragma unroll
      for (int v = 0; v < 8; ++v)
        pw[(v + 8 * half) * 40 + jt * 16 + lr] = f2bf(s[jt][v]);
    asm volatile("s_wait_dscnt 0x0" ::: "memory");

    bf16x16 ap = load_frag_lds(pw, 40, lr, half * 8, 16 + half * 8);
#pragma unroll
    for (int f = 0; f < 4; ++f) {
#pragma unroll
      for (int v = 0; v < 8; ++v) oacc[f][v] *= cs[v];
      bf16x16 bv = load_frag_lds(Vt, 40, f * 16 + lr, half * 8, 16 + half * 8);
      oacc[f] = wmma_bf16(ap, bv, oacc[f]);
    }
    __syncthreads();
  }

  // normalize, store bf16 O into [B*N, DIM] head slice
#pragma unroll
  for (int f = 0; f < 4; ++f)
#pragma unroll
    for (int v = 0; v < 8; ++v) {
      int row = row0 + v + 8 * half;
      int col = f * 16 + lr;
      o[(size_t)(b * NTOK + row) * DIM + hh * DHEAD + col] = f2bf(oacc[f][v] / lrow[v]);
    }
}

// ---------------- LayerNorm: one block per token, bf16 output ----------------
__global__ __launch_bounds__(256) void ln_kernel(const float* __restrict__ x,
                                                 const float* __restrict__ w,
                                                 const float* __restrict__ bshift,
                                                 unsigned short* __restrict__ y) {
  __shared__ float red[16];
  __shared__ float stats[2];
  const int m = blockIdx.x;
  const int t = threadIdx.x;
  const float* xr = x + (size_t)m * DIM;
  float vals[3];
  float s = 0.0f, s2 = 0.0f;
#pragma unroll
  for (int i = 0; i < 3; ++i) {
    float v = xr[t + i * 256];
    vals[i] = v;
    s += v; s2 += v * v;
  }
#pragma unroll
  for (int off = 16; off >= 1; off >>= 1) {
    s  += __shfl_xor(s,  off, 32);
    s2 += __shfl_xor(s2, off, 32);
  }
  const int lane = t & 31, wvi = t >> 5;
  if (lane == 0) { red[wvi] = s; red[8 + wvi] = s2; }
  __syncthreads();
  if (t == 0) {
    float a = 0.0f, a2 = 0.0f;
    for (int i = 0; i < 8; ++i) { a += red[i]; a2 += red[8 + i]; }
    float mu  = a * (1.0f / DIM);
    float var = a2 * (1.0f / DIM) - mu * mu;
    stats[0] = mu;
    stats[1] = rsqrtf(var + LN_EPS);
  }
  __syncthreads();
  const float mu = stats[0], rinv = stats[1];
#pragma unroll
  for (int i = 0; i < 3; ++i) {
    int d = t + i * 256;
    y[(size_t)m * DIM + d] = f2bf((vals[i] - mu) * rinv * w[d] + bshift[d]);
  }
}

// ---------------- weight/activation conversion kernels ----------------
// [L][K][N] fp32 -> [L][N][K] bf16 (coalesced writes)
__global__ __launch_bounds__(256) void convert_transpose(const float* __restrict__ src,
                                                         unsigned short* __restrict__ dst,
                                                         int K, int N) {
  int idx = blockIdx.x * 256 + threadIdx.x;
  int kn = K * N;
  int l = idx / kn;
  int rem = idx - l * kn;
  int n = rem / K;
  int k = rem - n * K;
  dst[idx] = f2bf(src[(size_t)l * kn + (size_t)k * N + n]);
}

__global__ __launch_bounds__(256) void convert_plain(const float* __restrict__ src,
                                                     unsigned short* __restrict__ dst) {
  int idx = blockIdx.x * 256 + threadIdx.x;
  dst[idx] = f2bf(src[idx]);
}

// c_f [B][F][N] fp32 -> xin [B*N][F] bf16
__global__ __launch_bounds__(256) void in_transpose(const float* __restrict__ cf,
                                                    unsigned short* __restrict__ xin) {
  int idx = blockIdx.x * 256 + threadIdx.x;
  int f = idx & (FDIM - 1);
  int m = idx >> 8;
  int b = m >> 10;
  int n = m & (NTOK - 1);
  xin[idx] = f2bf(cf[(size_t)(b * FDIM + f) * NTOK + n]);
}

// x [B*N][DIM] fp32 -> out [B][DIM][N]
__global__ __launch_bounds__(256) void out_transpose(const float* __restrict__ x,
                                                     float* __restrict__ out) {
  int idx = blockIdx.x * 256 + threadIdx.x;
  int n = idx & (NTOK - 1);
  int rest = idx >> 10;
  int d = rest % DIM;
  int b = rest / DIM;
  out[idx] = x[(size_t)(b * NTOK + n) * DIM + d];
}

// ---------------- host driver ----------------
extern "C" void kernel_launch(void* const* d_in, const int* in_sizes, int n_in,
                              void* d_out, int out_size, void* d_ws, size_t ws_size,
                              hipStream_t stream) {
  (void)in_sizes; (void)n_in; (void)out_size; (void)ws_size;
  const float* c_f    = (const float*)d_in[0];
  const float* conv_w = (const float*)d_in[1];   // [768,256] already [N][K]
  const float* conv_b = (const float*)d_in[2];
  const float* ln1_w  = (const float*)d_in[3];
  const float* ln1_b  = (const float*)d_in[4];
  const float* qkv_w  = (const float*)d_in[5];   // [8,768,2304]
  const float* out_w  = (const float*)d_in[6];   // [8,768,768]
  const float* out_b  = (const float*)d_in[7];
  const float* ln2_w  = (const float*)d_in[8];
  const float* ln2_b  = (const float*)d_in[9];
  const float* mlp_w1 = (const float*)d_in[10];  // [8,768,3072]
  const float* mlp_b1 = (const float*)d_in[11];
  const float* mlp_w2 = (const float*)d_in[12];  // [8,3072,768]
  const float* mlp_b2 = (const float*)d_in[13];

  // workspace layout (bf16 = unsigned short unless noted)
  char* cur = (char*)d_ws;
  auto take = [&cur](size_t bytes) { char* p = cur; cur += (bytes + 255) & ~(size_t)255; return p; };
  unsigned short* qkvT  = (unsigned short*)take((size_t)DEPTH * 3 * DIM * DIM * 2); // [8][2304][768]
  unsigned short* outT  = (unsigned short*)take((size_t)DEPTH * DIM * DIM * 2);     // [8][768][768]
  unsigned short* mlp1T = (unsigned short*)take((size_t)DEPTH * MLP * DIM * 2);     // [8][3072][768]
  unsigned short* mlp2T = (unsigned short*)take((size_t)DEPTH * DIM * MLP * 2);     // [8][768][3072]
  unsigned short* convB = (unsigned short*)take((size_t)DIM * FDIM * 2);            // [768][256]
  unsigned short* xin   = (unsigned short*)take((size_t)NROWS * FDIM * 2);
  float*          x     = (float*)take((size_t)NROWS * DIM * 4);
  unsigned short* y     = (unsigned short*)take((size_t)NROWS * DIM * 2);
  unsigned short* qkvb  = (unsigned short*)take((size_t)NROWS * 3 * DIM * 2);
  unsigned short* ob    = (unsigned short*)take((size_t)NROWS * DIM * 2);
  unsigned short* hb    = (unsigned short*)take((size_t)NROWS * MLP * 2);

  // one-time (per launch) weight conversion to bf16, transposed to [N][K]
  convert_transpose<<<(DEPTH * DIM * 3 * DIM) / 256, 256, 0, stream>>>(qkv_w, qkvT, DIM, 3 * DIM);
  convert_transpose<<<(DEPTH * DIM * DIM) / 256, 256, 0, stream>>>(out_w, outT, DIM, DIM);
  convert_transpose<<<(DEPTH * DIM * MLP) / 256, 256, 0, stream>>>(mlp_w1, mlp1T, DIM, MLP);
  convert_transpose<<<(DEPTH * MLP * DIM) / 256, 256, 0, stream>>>(mlp_w2, mlp2T, MLP, DIM);
  convert_plain<<<(DIM * FDIM) / 256, 256, 0, stream>>>(conv_w, convB);

  // tokens = conv1x1(c_f) + pos_embed  (fp32 residual stream x)
  in_transpose<<<(NROWS * FDIM) / 256, 256, 0, stream>>>(c_f, xin);
  gemm_kernel<EPI_BIAS_POS, false><<<dim3(DIM / 128, NROWS / 64), 256, 0, stream>>>(
      xin, convB, conv_b, nullptr, x, NROWS, DIM, FDIM, FDIM, DIM);

  for (int l = 0; l < DEPTH; ++l) {
    ln_kernel<<<NROWS, 256, 0, stream>>>(x, ln1_w + l * DIM, ln1_b + l * DIM, y);
    gemm_kernel<EPI_NONE, true><<<dim3(3 * DIM / 128, NROWS / 64), 256, 0, stream>>>(
        y, qkvT + (size_t)l * 3 * DIM * DIM, nullptr, nullptr, qkvb,
        NROWS, 3 * DIM, DIM, DIM, 3 * DIM);
    attn_kernel<<<dim3(NTOK / 128, HEADS, BATCH), 256, 0, stream>>>(qkvb, ob);
    gemm_kernel<EPI_BIAS_RESID, false><<<dim3(DIM / 128, NROWS / 64), 256, 0, stream>>>(
        ob, outT + (size_t)l * DIM * DIM, out_b + l * DIM, x, x,
        NROWS, DIM, DIM, DIM, DIM);
    ln_kernel<<<NROWS, 256, 0, stream>>>(x, ln2_w + l * DIM, ln2_b + l * DIM, y);
    gemm_kernel<EPI_BIAS_GELU, true><<<dim3(MLP / 128, NROWS / 64), 256, 0, stream>>>(
        y, mlp1T + (size_t)l * MLP * DIM, mlp_b1 + l * MLP, nullptr, hb,
        NROWS, MLP, DIM, DIM, MLP);
    gemm_kernel<EPI_BIAS_RESID, false><<<dim3(DIM / 128, NROWS / 64), 256, 0, stream>>>(
        hb, mlp2T + (size_t)l * DIM * MLP, mlp_b2 + l * DIM, x, x,
        NROWS, DIM, MLP, MLP, DIM);
  }

  out_transpose<<<(BATCH * DIM * NTOK) / 256, 256, 0, stream>>>(x, (float*)d_out);
}
